// EdgeDecoder_5488968204770
// MI455X (gfx1250) — matrix operand order repaired and checked by reading, not compiled, not run on hardware
//
#include <hip/hip_runtime.h>

typedef __attribute__((ext_vector_type(16))) __bf16       v16bf;
typedef __attribute__((ext_vector_type(8)))  float        v8f;
typedef __attribute__((ext_vector_type(8)))  unsigned int uint8v;

// f32x2 -> packed bf16x2 (round-half-up), one v_add per half + one v_perm_b32.
__device__ __forceinline__ unsigned int pack2_bf16(float lo, float hi) {
    unsigned int ulo = __builtin_bit_cast(unsigned int, lo) + 0x8000u;
    unsigned int uhi = __builtin_bit_cast(unsigned int, hi) + 0x8000u;
    // dst bytes: {uhi.b3, uhi.b2, ulo.b3, ulo.b2}  ({s0,s1}: s1=bytes0-3)
    return __builtin_amdgcn_perm(uhi, ulo, 0x07060302u);
}

// ---------------------------------------------------------------------------
// Prep: swizzle W1 (256x128 f32, row-major) for both relations into bf16
// B-fragments in workspace.
// Fragment (kt, nt) layout: wsB[r*16384 + ((kt*8+nt)*32 + lane)*8 + j]
//   lane: n = nt*16 + (lane&15), h = lane>>4
//   dword j holds bf16 pair (K0, K0+1), K0 = kt*32 + h*16 + 2*j
// ---------------------------------------------------------------------------
__global__ __launch_bounds__(256) void edgemlp_prep_w1(
    const float* __restrict__ w1_gd, const float* __restrict__ w1_gg,
    unsigned int* __restrict__ wsB)
{
    int tid = blockIdx.x * blockDim.x + threadIdx.x;   // 0 .. 32767
    if (tid >= 2 * 16384) return;
    int r  = tid >> 14;
    int t  = tid & 16383;
    int j  = t & 7;
    int L  = (t >> 3) & 31;
    int f  = t >> 8;            // 0..63
    int kt = f >> 3;
    int nt = f & 7;
    int n  = nt * 16 + (L & 15);
    int K0 = kt * 32 + (L >> 4) * 16 + 2 * j;
    const float* w1 = r ? w1_gg : w1_gd;
    float lo = w1[(size_t)K0 * 128 + n];
    float hi = w1[(size_t)(K0 + 1) * 128 + n];
    wsB[tid] = pack2_bf16(lo, hi);
}

// Gather one 16-edge A tile (8 x v16bf fragments), A 16-bit 16x32 layout.
__device__ __forceinline__ void gather_a_tile(
    const float* __restrict__ rs, const float* __restrict__ rd,
    int h, v16bf a_v[8])
{
#pragma unroll
    for (int kt = 0; kt < 8; ++kt) {
        uint8v u;
#pragma unroll
        for (int jp = 0; jp < 4; ++jp) {
            int kc = h * 16 + 4 * jp;
            int k  = kt * 32 + kc;                              // mult of 4, in-half
            const float* p = (k < 128) ? (rs + k) : (rd + (k - 128));
            float4 pv = *(const float4*)p;                      // 16B gather
            u[2 * jp]     = pack2_bf16(pv.x, pv.y);
            u[2 * jp + 1] = pack2_bf16(pv.z, pv.w);
        }
        a_v[kt] = __builtin_bit_cast(v16bf, u);
    }
}

// ---------------------------------------------------------------------------
// Main: blockIdx.y = relation (block-uniform -> scalar selects).
// Block = 8 wave32; B fragments staged once into 64KB LDS and shared by all
// waves. One wave = 32 edges (two 16-row A sub-tiles); each B fragment feeds
// two wmmas; LDS reads are double-buffered one fragment ahead.
// ---------------------------------------------------------------------------
__global__ __launch_bounds__(256) void edgemlp_main(
    const float* __restrict__ zg, const float* __restrict__ zd,
    const int* __restrict__ e_gd, const int* __restrict__ e_gg,
    const float* __restrict__ b1_gd, const float* __restrict__ w2_gd,
    const float* __restrict__ b2_gd,
    const float* __restrict__ b1_gg, const float* __restrict__ w2_gg,
    const float* __restrict__ b2_gg,
    const unsigned int* __restrict__ wsB,
    float* __restrict__ out, int E, int tilesPerRel)
{
    extern __shared__ unsigned int ldsB[];      // 16384 dwords = 64KB

    const int r   = blockIdx.y;                 // block-uniform relation
    const int tid = threadIdx.x;

    // ---- Stage this relation's B fragments into LDS (coalesced b128s) ----
    const unsigned int* wbg = wsB + (size_t)r * 16384;
#pragma unroll
    for (int q = 0; q < 16; ++q) {
        int idx = (q * 256 + tid) * 4;
        *(uint4*)(&ldsB[idx]) = *(const uint4*)(&wbg[idx]);
    }
    __syncthreads();

    const int lane = tid & 31;
    const int waveInBlk = tid >> 5;
    const int tile = blockIdx.x * 8 + waveInBlk;      // 32-edge tile in relation
    if (tile >= tilesPerRel) return;                   // wave-uniform

    const int m = lane & 15;        // A row / C col-within-tile
    const int h = lane >> 4;        // half of wave

    const int*   edges = r ? e_gg : e_gd;              // scalar selects
    const float* zdst  = r ? zg : zd;

    // Two edges per lane: sub-tile 0 -> rows 0..15, sub-tile 1 -> rows 16..31
    long e0 = (long)tile * 32 + m;        if (e0 >= E) e0 = E - 1;
    long e1 = (long)tile * 32 + 16 + m;   if (e1 >= E) e1 = E - 1;
    const float* rs0 = zg   + (size_t)edges[e0] * 128;
    const float* rd0 = zdst + (size_t)edges[(size_t)E + e0] * 128;
    const float* rs1 = zg   + (size_t)edges[e1] * 128;
    const float* rd1 = zdst + (size_t)edges[(size_t)E + e1] * 128;

    v16bf a0[8], a1[8];
    gather_a_tile(rs0, rd0, h, a0);
    gather_a_tile(rs1, rd1, h, a1);

    const float* b1  = r ? b1_gg : b1_gd;
    const float* w2  = r ? w2_gg : w2_gd;
    const float  b2s = (r ? b2_gg : b2_gd)[0];

    const unsigned int* lb = ldsB + (lane << 3);       // one addr reg + imm offs

    float acc0[8], acc1[8];
#pragma unroll
    for (int v = 0; v < 8; ++v) { acc0[v] = 0.0f; acc1[v] = 0.0f; }

#pragma unroll
    for (int nt = 0; nt < 8; ++nt) {
        v8f c0 = {}, c1 = {};
        // double-buffered LDS fragment stream: load kt+1 before wmma on kt
        uint8v bu = *(const uint8v*)(lb + (nt << 8));          // kt = 0
#pragma unroll
        for (int kt = 0; kt < 8; ++kt) {
            uint8v bnx = bu;
            if (kt < 7)
                bnx = *(const uint8v*)(lb + (((kt + 1) * 8 + nt) << 8));
            v16bf b = __builtin_bit_cast(v16bf, bu);
            c0 = __builtin_amdgcn_wmma_f32_16x16x32_bf16(
                     false, a0[kt], false, b, (short)0, c0, false, false);
            c1 = __builtin_amdgcn_wmma_f32_16x16x32_bf16(
                     false, a1[kt], false, b, (short)0, c1, false, false);
            bu = bnx;
        }
        const int n    = nt * 16 + m;
        const float bn = b1[n];
        const float wn = w2[n];
#pragma unroll
        for (int v = 0; v < 8; ++v) {          // C: VGPR v -> row v+8h, col n
            float h0 = fmaxf(c0[v] + bn, 0.0f);
            float h1 = fmaxf(c1[v] + bn, 0.0f);
            acc0[v] += h0 * wn;
            acc1[v] += h1 * wn;
        }
    }

    // Reduce across the 16 lanes of each half
#pragma unroll
    for (int v = 0; v < 8; ++v) {
#pragma unroll
        for (int off = 1; off < 16; off <<= 1) {
            acc0[v] += __shfl_xor(acc0[v], off, 32);
            acc1[v] += __shfl_xor(acc1[v], off, 32);
        }
    }

    if (m == 0) {                               // lane0 -> rows +0..7, lane16 -> +8..15
#pragma unroll
        for (int v = 0; v < 8; ++v) {
            long eo0 = (long)tile * 32 + v + 8 * h;
            long eo1 = eo0 + 16;
            if (eo0 < E) out[(size_t)r * E + eo0] = acc0[v] + b2s;
            if (eo1 < E) out[(size_t)r * E + eo1] = acc1[v] + b2s;
        }
    }
}

extern "C" void kernel_launch(void* const* d_in, const int* in_sizes, int n_in,
                              void* d_out, int out_size, void* d_ws, size_t ws_size,
                              hipStream_t stream) {
    const float* zg    = (const float*)d_in[0];
    const float* zd    = (const float*)d_in[1];
    const int*   e_gd  = (const int*)d_in[2];
    const int*   e_gg  = (const int*)d_in[3];
    const float* w1_gd = (const float*)d_in[4];
    const float* b1_gd = (const float*)d_in[5];
    const float* w2_gd = (const float*)d_in[6];
    const float* b2_gd = (const float*)d_in[7];
    const float* w1_gg = (const float*)d_in[8];
    const float* b1_gg = (const float*)d_in[9];
    const float* w2_gg = (const float*)d_in[10];
    const float* b2_gg = (const float*)d_in[11];

    unsigned int* wsB = (unsigned int*)d_ws;   // needs 2*16384*4 = 128KB
    float* out = (float*)d_out;

    const int E = in_sizes[2] / 2;             // edges arrays are [2, E]
    const int tilesPerRel = (E + 31) / 32;     // 32 edges per wave
    const int blocksPerRel = (tilesPerRel + 7) / 8;   // 8 waves per block

    edgemlp_prep_w1<<<(2 * 16384 + 255) / 256, 256, 0, stream>>>(w1_gd, w1_gg, wsB);
    edgemlp_main<<<dim3(blocksPerRel, 2), 256, 65536, stream>>>(
        zg, zd, e_gd, e_gg,
        b1_gd, w2_gd, b2_gd,
        b1_gg, w2_gg, b2_gg,
        wsB, out, E, tilesPerRel);
}
